// AttnBlock3d_25769804079
// MI455X (gfx1250) — matrix operand order repaired and checked by reading, not compiled, not run on hardware
//
#include <hip/hip_runtime.h>
#include <hip/hip_bf16.h>

// ---------------------------------------------------------------------------
// AttnBlock3d on gfx1250 (MI455X): GroupNorm + QKV proj + flash-attention +
// out-proj + residual.  All GEMMs via v_wmma_f32_16x16x32_bf16 (wave32).
// B=2, C=256, G=32, N=4096.
// ---------------------------------------------------------------------------

#define Bsz 2
#define Cch 256
#define Ggr 32
#define Nsp 4096
#define EPS 1e-5f
#define QSCALE 0.0625f   // 256^-0.5

typedef __attribute__((ext_vector_type(16))) __bf16 v16bf;
typedef __attribute__((ext_vector_type(8)))  __bf16 v8bf;
typedef __attribute__((ext_vector_type(8)))  float  v8f;

__device__ __forceinline__ v8f wmma_bf16(v16bf a, v16bf b, v8f c) {
  // D = A(16x32 bf16) * B(32x16 bf16) + C(16x16 f32)
  return __builtin_amdgcn_wmma_f32_16x16x32_bf16(
      /*neg_a=*/false, a, /*neg_b=*/false, b,
      /*c_mod=*/(short)0, c, /*reuse_a=*/false, /*reuse_b=*/false);
}

// A-matrix fragment (16x32, M rows, K cols). Lane L: row = L&15.
// elements 0..15 -> K = kbase + (L>=16)*16 + 0..15 (contiguous).
// Caller passes p already offset by row*ld + kbase + (L>=16)*16.
__device__ __forceinline__ v16bf frag_a(const __bf16* p) {
  v8bf c0 = *(const v8bf*)p;
  v8bf c1 = *(const v8bf*)(p + 8);
  v16bf r;
#pragma unroll
  for (int i = 0; i < 8; ++i) { r[i] = c0[i]; r[i + 8] = c1[i]; }
  return r;
}

// B-matrix fragment (32x16, K rows, N cols). Lane L: col = L&15.
// elements 0..15 -> K = kbase + (L>=16)*16 + 0..15 (contiguous).
// Caller passes p already offset by col*ld + kbase + (L>=16)*16.
__device__ __forceinline__ v16bf frag_b(const __bf16* p) {
  v8bf c0 = *(const v8bf*)p;
  v8bf c1 = *(const v8bf*)(p + 8);
  v16bf r;
#pragma unroll
  for (int i = 0; i < 8; ++i) { r[i] = c0[i]; r[i + 8] = c1[i]; }
  return r;
}

// ---------------------------------------------------------------------------
// 1) GroupNorm -> hT [B][N][C] bf16
// ---------------------------------------------------------------------------
__global__ __launch_bounds__(256) void gn_kernel(const float* __restrict__ x,
                                                 const float* __restrict__ gamma,
                                                 const float* __restrict__ beta,
                                                 __bf16* __restrict__ hT) {
  __shared__ float sred[2][8];
  const int b = blockIdx.x >> 5;
  const int g = blockIdx.x & 31;
  const float* xp = x + ((size_t)b * Cch + g * 8) * Nsp;  // 8 channels x N
  float s = 0.f, ss = 0.f;
  for (int i = threadIdx.x; i < 8 * Nsp; i += 256) {
    float v = xp[i];
    s += v; ss += v * v;
  }
#pragma unroll
  for (int m = 1; m < 32; m <<= 1) {
    s  += __shfl_xor(s,  m, 32);
    ss += __shfl_xor(ss, m, 32);
  }
  if ((threadIdx.x & 31) == 0) {
    sred[0][threadIdx.x >> 5] = s;
    sred[1][threadIdx.x >> 5] = ss;
  }
  __syncthreads();
  s = 0.f; ss = 0.f;
#pragma unroll
  for (int i = 0; i < 8; ++i) { s += sred[0][i]; ss += sred[1][i]; }
  const float inv = 1.0f / (8.0f * Nsp);
  const float mu = s * inv;
  const float var = ss * inv - mu * mu;
  const float rstd = rsqrtf(var + EPS);
  for (int i = threadIdx.x; i < 8 * Nsp; i += 256) {
    const int cl = i >> 12;        // i / 4096
    const int n  = i & (Nsp - 1);
    const int c  = g * 8 + cl;
    float v = (xp[i] - mu) * rstd * gamma[c] + beta[c];
    hT[((size_t)b * Nsp + n) * Cch + c] = (__bf16)v;
  }
}

// ---------------------------------------------------------------------------
// 2) f32 -> bf16 weight conversion
// ---------------------------------------------------------------------------
__global__ void wcvt_kernel(const float* __restrict__ src, __bf16* __restrict__ dst, int count) {
  int i = blockIdx.x * blockDim.x + threadIdx.x;
  if (i < count) dst[i] = (__bf16)src[i];
}

// ---------------------------------------------------------------------------
// 3) Projection: D(o,n) = W(o,:) . h(:,n) + bias[o]
//    mode 0: q -> [N][C] scaled by QSCALE
//    mode 1: k -> [N][C]
//    mode 2: v -> [C][N]
// One 16x16 output tile per wave; K = 256 in 8 steps of 32.
// ---------------------------------------------------------------------------
__global__ __launch_bounds__(256) void proj_kernel(const __bf16* __restrict__ W,
                                                   const float* __restrict__ bias,
                                                   const __bf16* __restrict__ hT,
                                                   __bf16* __restrict__ dst,
                                                   int mode) {
  const int wid  = (blockIdx.x * blockDim.x + threadIdx.x) >> 5;
  const int lane = threadIdx.x & 31;
  const int b  = wid >> 12;        // 4096 tiles per batch
  const int rem = wid & 4095;
  const int ot = rem >> 8;         // 0..15  (C/16)
  const int nt = rem & 255;        // 0..255 (N/16)
  const int hi = lane >> 4, lo16 = lane & 15;
  const __bf16* hb = hT + (size_t)b * Nsp * Cch;

  v8f acc;
#pragma unroll
  for (int r = 0; r < 8; ++r) acc[r] = 0.f;

#pragma unroll
  for (int kk = 0; kk < 8; ++kk) {
    v16bf aW = frag_a(W  + (size_t)(ot * 16 + lo16) * Cch + kk * 32 + hi * 16);
    v16bf bh = frag_b(hb + (size_t)(nt * 16 + lo16) * Cch + kk * 32 + hi * 16);
    acc = wmma_bf16(aW, bh, acc);
  }

#pragma unroll
  for (int r = 0; r < 8; ++r) {
    const int o = ot * 16 + r + 8 * hi;   // D row
    const int n = nt * 16 + lo16;         // D col
    float val = acc[r] + bias[o];
    if (mode == 0) val *= QSCALE;
    if (mode == 2)
      dst[((size_t)b * Cch + o) * Nsp + n] = (__bf16)val;
    else
      dst[((size_t)b * Nsp + n) * Cch + o] = (__bf16)val;
  }
}

// ---------------------------------------------------------------------------
// 4) Flash attention: one wave per 16 query rows.
//    q,k: [B][N][C] bf16 (q prescaled), v: [B][C][N] bf16 -> a: [B][N][C] bf16
// ---------------------------------------------------------------------------
__global__ __launch_bounds__(32) void attn_kernel(const __bf16* __restrict__ q,
                                                  const __bf16* __restrict__ k,
                                                  const __bf16* __restrict__ v,
                                                  __bf16* __restrict__ a) {
  __shared__ __bf16 pbuf[16 * 32];   // P tile [n=16][m=32], rows 64B
  const int lane = threadIdx.x;
  const int id = blockIdx.x;
  const int b = id >> 8;
  const int nbase = (id & 255) << 4;
  const int hi = lane >> 4, lo16 = lane & 15;
  const __bf16* qb = q + (size_t)b * Nsp * Cch;
  const __bf16* kb = k + (size_t)b * Nsp * Cch;
  const __bf16* vb = v + (size_t)b * Cch * Nsp;

  // Resident Q A-fragments: 8 K-steps of 32 (rows nbase..nbase+15).
  v16bf qa[8];
#pragma unroll
  for (int kk = 0; kk < 8; ++kk)
    qa[kk] = frag_a(qb + (size_t)(nbase + lo16) * Cch + kk * 32 + hi * 16);

  // Output accumulators: 16 column tiles of C=256, f32.
  v8f acc[16];
#pragma unroll
  for (int ct = 0; ct < 16; ++ct)
#pragma unroll
    for (int r = 0; r < 8; ++r) acc[ct][r] = 0.f;

  float mrow[8], lrow[8];
#pragma unroll
  for (int r = 0; r < 8; ++r) { mrow[r] = -1e30f; lrow[r] = 0.f; }

  for (int mt = 0; mt < Nsp; mt += 32) {
    // ---- S = Qblk(16x256) @ K(256 x [mt..mt+32)) : two 16x16 f32 tiles ----
    v8f S0, S1;
#pragma unroll
    for (int r = 0; r < 8; ++r) { S0[r] = 0.f; S1[r] = 0.f; }
#pragma unroll
    for (int kk = 0; kk < 8; ++kk) {
      v16bf b0 = frag_b(kb + (size_t)(mt      + lo16) * Cch + kk * 32 + hi * 16);
      v16bf b1 = frag_b(kb + (size_t)(mt + 16 + lo16) * Cch + kk * 32 + hi * 16);
      S0 = wmma_bf16(qa[kk], b0, S0);
      S1 = wmma_bf16(qa[kk], b1, S1);
    }

    // ---- online softmax (rows live in lanes of the same 16-lane half) ----
    float p0[8], p1[8], alpha[8];
#pragma unroll
    for (int r = 0; r < 8; ++r) {
      float t = fmaxf(S0[r], S1[r]);
      t = fmaxf(t, __shfl_xor(t, 1, 32));
      t = fmaxf(t, __shfl_xor(t, 2, 32));
      t = fmaxf(t, __shfl_xor(t, 4, 32));
      t = fmaxf(t, __shfl_xor(t, 8, 32));
      const float mnew = fmaxf(mrow[r], t);
      alpha[r] = __expf(mrow[r] - mnew);
      mrow[r] = mnew;
      p0[r] = __expf(S0[r] - mnew);
      p1[r] = __expf(S1[r] - mnew);
      float s = p0[r] + p1[r];
      s += __shfl_xor(s, 1, 32);
      s += __shfl_xor(s, 2, 32);
      s += __shfl_xor(s, 4, 32);
      s += __shfl_xor(s, 8, 32);
      lrow[r] = lrow[r] * alpha[r] + s;
    }
#pragma unroll
    for (int ct = 0; ct < 16; ++ct)
#pragma unroll
      for (int r = 0; r < 8; ++r) acc[ct][r] *= alpha[r];

    // ---- stage P (C/D layout) through LDS into A layout ----
    __syncthreads();
#pragma unroll
    for (int r = 0; r < 8; ++r) {
      pbuf[(r + 8 * hi) * 32 + lo16]      = (__bf16)p0[r];
      pbuf[(r + 8 * hi) * 32 + 16 + lo16] = (__bf16)p1[r];
    }
    __syncthreads();
    v16bf pa = frag_a(&pbuf[lo16 * 32 + hi * 16]);

    // ---- acc += P(16x32) @ V([mt..mt+32) x 256) ----
#pragma unroll
    for (int ct = 0; ct < 16; ++ct) {
      v16bf bv = frag_b(vb + (size_t)(ct * 16 + lo16) * Nsp + mt + hi * 16);
      acc[ct] = wmma_bf16(pa, bv, acc[ct]);
    }
  }

  // ---- normalize and store a [B][N][C] bf16 ----
#pragma unroll
  for (int r = 0; r < 8; ++r) lrow[r] = 1.0f / lrow[r];
#pragma unroll
  for (int ct = 0; ct < 16; ++ct)
#pragma unroll
    for (int r = 0; r < 8; ++r) {
      const float val = acc[ct][r] * lrow[r];
      a[((size_t)b * Nsp + nbase + r + 8 * hi) * Cch + ct * 16 + lo16] = (__bf16)val;
    }
}

// ---------------------------------------------------------------------------
// 5) Output projection + residual: out[b,o,n] = x + Wp(o,:).a(n,:) + bp[o]
//    D tile: M rows = n, N cols = o.  B operand = Wp row-major (o rows give
//    B(c,o) = Wp[o][c], K-contiguous).
// ---------------------------------------------------------------------------
__global__ __launch_bounds__(256) void outproj_kernel(const __bf16* __restrict__ aT,
                                                      const __bf16* __restrict__ WpB,
                                                      const float* __restrict__ bp,
                                                      const float* __restrict__ x,
                                                      float* __restrict__ out) {
  const int wid  = (blockIdx.x * blockDim.x + threadIdx.x) >> 5;
  const int lane = threadIdx.x & 31;
  const int b  = wid >> 12;
  const int rem = wid & 4095;
  const int nt = rem >> 4;       // 0..255
  const int ot = rem & 15;       // 0..15
  const int hi = lane >> 4, lo16 = lane & 15;
  const __bf16* ab = aT + (size_t)b * Nsp * Cch;

  v8f acc;
#pragma unroll
  for (int r = 0; r < 8; ++r) acc[r] = 0.f;

#pragma unroll
  for (int kk = 0; kk < 8; ++kk) {
    v16bf aa = frag_a(ab  + (size_t)(nt * 16 + lo16) * Cch + kk * 32 + hi * 16);
    v16bf bw = frag_b(WpB + (size_t)(ot * 16 + lo16) * Cch + kk * 32 + hi * 16);
    acc = wmma_bf16(aa, bw, acc);
  }

#pragma unroll
  for (int r = 0; r < 8; ++r) {
    const int n = nt * 16 + r + 8 * hi;   // D row
    const int o = ot * 16 + lo16;         // D col
    const size_t idx = ((size_t)b * Cch + o) * Nsp + n;
    out[idx] = x[idx] + bp[o] + acc[r];
  }
}

// ---------------------------------------------------------------------------
// Host-side orchestration
// ---------------------------------------------------------------------------
extern "C" void kernel_launch(void* const* d_in, const int* in_sizes, int n_in,
                              void* d_out, int out_size, void* d_ws, size_t ws_size,
                              hipStream_t stream) {
  const float* x     = (const float*)d_in[0];
  const float* gamma = (const float*)d_in[1];
  const float* beta  = (const float*)d_in[2];
  const float* Wq    = (const float*)d_in[3];
  const float* bq    = (const float*)d_in[4];
  const float* Wk    = (const float*)d_in[5];
  const float* bk    = (const float*)d_in[6];
  const float* Wv    = (const float*)d_in[7];
  const float* bv    = (const float*)d_in[8];
  const float* Wp    = (const float*)d_in[9];
  const float* bp    = (const float*)d_in[10];
  float* out = (float*)d_out;

  // Carve workspace (256B-aligned slices).
  char* ws = (char*)d_ws;
  size_t off = 0;
  auto take = [&](size_t bytes) -> char* {
    char* p = ws + off;
    off += (bytes + 255) & ~(size_t)255;
    return p;
  };
  const size_t actBytes = (size_t)Bsz * Nsp * Cch * sizeof(__bf16);  // 4 MB each
  __bf16* hT  = (__bf16*)take(actBytes);   // [B][N][C]
  __bf16* qT  = (__bf16*)take(actBytes);   // [B][N][C], prescaled
  __bf16* kT  = (__bf16*)take(actBytes);   // [B][N][C]
  __bf16* vCN = (__bf16*)take(actBytes);   // [B][C][N]
  __bf16* aT  = (__bf16*)take(actBytes);   // [B][N][C]
  const size_t wBytes = (size_t)Cch * Cch * sizeof(__bf16);
  __bf16* wqb = (__bf16*)take(wBytes);
  __bf16* wkb = (__bf16*)take(wBytes);
  __bf16* wvb = (__bf16*)take(wBytes);
  __bf16* wpb = (__bf16*)take(wBytes);

  // 1) GroupNorm -> hT bf16
  gn_kernel<<<Bsz * Ggr, 256, 0, stream>>>(x, gamma, beta, hT);

  // 2) Weight conversions
  const int wcnt = Cch * Cch;
  wcvt_kernel<<<wcnt / 256, 256, 0, stream>>>(Wq, wqb, wcnt);
  wcvt_kernel<<<wcnt / 256, 256, 0, stream>>>(Wk, wkb, wcnt);
  wcvt_kernel<<<wcnt / 256, 256, 0, stream>>>(Wv, wvb, wcnt);
  wcvt_kernel<<<wcnt / 256, 256, 0, stream>>>(Wp, wpb, wcnt);

  // 3) Projections: 8192 tiles (16x16) -> 8 waves per 256-thread block
  const int projBlocks = (Bsz * (Cch / 16) * (Nsp / 16)) / 8;  // 1024
  proj_kernel<<<projBlocks, 256, 0, stream>>>(wqb, bq, hT, qT, 0);
  proj_kernel<<<projBlocks, 256, 0, stream>>>(wkb, bk, hT, kT, 1);
  proj_kernel<<<projBlocks, 256, 0, stream>>>(wvb, bv, hT, vCN, 2);

  // 4) Flash attention: one wave per 16 query rows
  attn_kernel<<<Bsz * (Nsp / 16), 32, 0, stream>>>(qT, kT, vCN, aT);

  // 5) Output projection + residual
  outproj_kernel<<<projBlocks, 256, 0, stream>>>(aT, wpb, bp, x, out);
}
